// MNIST_Model_3_levels_w_Att_10565619549012
// MI455X (gfx1250) — compile-verified, matrix operand order checked by reading, not run on hardware
//
#include <hip/hip_runtime.h>
#include <hip/hip_bf16.h>
#include <math.h>
#include <stdint.h>

// ---------------------------------------------------------------------------
// MNIST 3-level attention MIL — CDNA5 (gfx1250) WMMA implementation.
// N=16384 instances, B1=1024, B2=64 bags, conv1 5x5x1x32, conv2 5x5x32x64,
// D=1024, ATT=256, CLS=512.  All big GEMMs via v_wmma_f32_16x16x32_f16;
// operands move through LDS with b128 accesses; weight tiles streamed by the
// Tensor Data Mover (double-buffered) when the toolchain exposes it.
// ---------------------------------------------------------------------------

typedef __attribute__((ext_vector_type(16))) _Float16 v16h;
typedef __attribute__((ext_vector_type(8)))  _Float16 h8;
typedef __attribute__((ext_vector_type(2)))  _Float16 h2f;
typedef __attribute__((ext_vector_type(8)))  float    v8f;

#define N_INST 16384
#define NBAG1  1024
#define NBAG2  64

#if (__has_builtin(__builtin_amdgcn_tensor_load_to_lds) && \
     __has_builtin(__builtin_amdgcn_s_wait_tensorcnt))
#define HAVE_TDM 1
#pragma message("CDNA5 probe: TDM tensor_load_to_lds path ENABLED")
#else
#define HAVE_TDM 0
#pragma message("CDNA5 probe: TDM builtin NOT available - cooperative-copy fallback")
#endif

static __device__ __forceinline__ v8f wmma16(v16h a, v16h b, v8f c) {
  // D = A(16x32 f16) * B(32x16 f16) + C(16x16 f32)
  return __builtin_amdgcn_wmma_f32_16x16x32_f16(
      false, a, false, b, (short)0, c, false, false);
}

// Fragment slot s (0..15) -> K:  lanes 0-15: {0..7,16..23}, lanes 16-31:
// {8..15,24..31}.  With K-contiguous operand storage this is two runs of 8
// consecutive halfs at offsets laneHi*8 and 16+laneHi*8 -> two b128 loads.
static __device__ __forceinline__ v16h mkfrag(h8 lo, h8 hi) {
  v16h r;
#pragma unroll
  for (int s = 0; s < 8; ++s) { r[s] = lo[s]; r[s + 8] = hi[s]; }
  return r;
}

static __device__ __forceinline__ v16h load_frag(const _Float16* kcol, int laneHi) {
  h8 lo = *(const h8*)(kcol + laneHi * 8);
  h8 hi = *(const h8*)(kcol + 16 + laneHi * 8);
  return mkfrag(lo, hi);
}

// ------------------------- Tensor Data Mover ------------------------------
#if HAVE_TDM
typedef unsigned int v4u __attribute__((ext_vector_type(4)));
typedef int          v8i __attribute__((ext_vector_type(8)));
typedef int          v4i __attribute__((ext_vector_type(4)));

// 1-D f16 tile: copy `nelem` halfs from global `gsrc` into LDS `ldsAddr`.
// D# group0: count=1 | lds_addr | global_addr | type=2 ("image").
// D# group1: data_size=2B, tensor_dim0=nelem, tensor_dim1=1, tile=nelem x 1.
static __device__ __forceinline__ void tdm_load_f16_1d(
    unsigned int ldsAddr, const _Float16* gsrc, unsigned int nelem) {
  unsigned long long ga = (unsigned long long)(uintptr_t)gsrc;
  v4u g0;
  g0[0] = 1u;                                         // count=1 (valid)
  g0[1] = ldsAddr;                                    // lds_addr
  g0[2] = (unsigned int)(ga & 0xffffffffu);           // global_addr[31:0]
  g0[3] = (unsigned int)((ga >> 32) & 0x01ffffffu)    // global_addr[56:32]
          | (2u << 30);                               // type=2
  v8i g1;
  g1[0] = (int)(1u << 16);                            // data_size=1 -> 2 bytes
  g1[1] = (int)(nelem << 16);                         // tensor_dim0[15:0]
  g1[2] = (int)((nelem >> 16) & 0xffffu)              // tensor_dim0[31:16]
          | (int)(1u << 16);                          // tensor_dim1[15:0]=1
  g1[3] = (int)(nelem << 16);                         // tile_dim0 = nelem
  g1[4] = 1;                                          // tile_dim1 = 1
  g1[5] = (int)nelem;                                 // tensor_dim0_stride lo
  g1[6] = 0;
  g1[7] = 0;
  v4i g2 = {0, 0, 0, 0};
  v4i g3 = {0, 0, 0, 0};
#if __clang_major__ >= 23
  v8i gx = {0, 0, 0, 0, 0, 0, 0, 0};
  __builtin_amdgcn_tensor_load_to_lds(g0, g1, g2, g3, gx, 0);
#else
  __builtin_amdgcn_tensor_load_to_lds(g0, g1, g2, g3, 0);
#endif
}
#endif

// ------------------------- weight packing ---------------------------------
// All weights stored K-contiguous per output column (transposed) so WMMA B
// fragments are two b128 loads per lane.

// conv1 [5,5,1,32] -> W1t[n=32][k=32] (taps 25..31 zero)
__global__ void k_pack_w1(const float* __restrict__ w, _Float16* __restrict__ dst) {
  int i = threadIdx.x;                    // 1 block x 1024
  int n = i >> 5, k = i & 31;
  dst[i] = (k < 25) ? (_Float16)w[k * 32 + n] : (_Float16)0.f;
}

// conv2 [5,5,32,64]: per-tap transpose -> W2t[t=25][co=64][ci=32]
__global__ void k_pack_w2(const float* __restrict__ w, _Float16* __restrict__ dst) {
  int i = blockIdx.x * 256 + threadIdx.x;
  if (i < 51200) {
    int t = i >> 11, r = i & 2047;
    int co = r >> 5, ci = r & 31;
    dst[i] = (_Float16)w[(t * 32 + ci) * 64 + co];
  }
}

// att W [1024,256] -> AWt[n=256][k=1024]
__global__ void k_pack_att(const float* __restrict__ w, _Float16* __restrict__ dst) {
  int i = blockIdx.x * 256 + threadIdx.x;
  if (i < 262144) {
    int n = i >> 10, k = i & 1023;
    dst[i] = (_Float16)w[k * 256 + n];
  }
}

// ------------------------- conv1 + pool -----------------------------------
// per-image implicit GEMM: M=576 (24x24), N=32, K=32 (25 taps zero-padded).
// Conv output kept channel-major in LDS so each C tile writes two b128s.
__global__ __launch_bounds__(256)
void k_conv1(const float* __restrict__ x, const _Float16* __restrict__ W1t,
             const float* __restrict__ b1, _Float16* __restrict__ h1) {
  __shared__ _Float16 img[28 * 32];         // zero-padded tail (OOB taps hit 0)
  __shared__ _Float16 co[32 * 576];         // [channel][position]
  const int imgId  = blockIdx.x;
  const int tid    = threadIdx.x;
  const int lane   = tid & 31;
  const int wave   = tid >> 5;
  const int laneHi = lane >> 4;
  const int laneLo = lane & 15;

  const float* xp = x + (size_t)imgId * 784;
  for (int i = tid; i < 28 * 32; i += 256)
    img[i] = (i < 784) ? (_Float16)xp[i] : (_Float16)0.f;

  // B fragments for both channel tiles: two b128 global loads each
  v16h bfrag[2];
#pragma unroll
  for (int nt = 0; nt < 2; ++nt)
    bfrag[nt] = load_frag(W1t + (nt * 16 + laneLo) * 32, laneHi);
  __syncthreads();

  for (int mt = wave; mt < 36; mt += 8) {
    int p  = mt * 16 + laneLo;              // output position 0..575
    int oy = p / 24, ox = p % 24;
    v16h a;
#pragma unroll
    for (int s = 0; s < 16; ++s) {
      int kk = (s < 8 ? s : s + 8) + laneHi * 8;
      int ky = kk / 5, kx = kk - 5 * ky;    // taps >=25 have zero weights in W1t
      a[s] = img[(oy + ky) * 28 + (ox + kx)];
    }
#pragma unroll
    for (int nt = 0; nt < 2; ++nt) {
      v8f c = {};
      c = wmma16(a, bfrag[nt], c);
      int ch = nt * 16 + laneLo;
      float bias = b1[ch];
      h8 st;
#pragma unroll
      for (int r = 0; r < 8; ++r)           // 8 consecutive positions
        st[r] = (_Float16)fmaxf(c[r] + bias, 0.f);
      *(h8*)(&co[ch * 576 + mt * 16 + laneHi * 8]) = st;  // ds_store_b128
    }
  }
  __syncthreads();

  // 2x2 maxpool: 24x24x32 -> 12x12x32; process channel pairs -> b32 stores
  _Float16* out = h1 + (size_t)imgId * 4608;
  for (int o2 = tid * 2; o2 < 4608; o2 += 512) {
    int ch = o2 & 31, pix = o2 >> 5;        // ch even; o2+1 -> ch+1, same pixel
    int px = pix % 12, py = pix / 12;
    int p0 = (py * 2) * 24 + px * 2;
    h2f a0 = *(const h2f*)(co + ch * 576 + p0);
    h2f a1 = *(const h2f*)(co + ch * 576 + p0 + 24);
    h2f b0 = *(const h2f*)(co + (ch + 1) * 576 + p0);
    h2f b1 = *(const h2f*)(co + (ch + 1) * 576 + p0 + 24);
    float m0 = fmaxf(fmaxf((float)a0[0], (float)a0[1]),
                     fmaxf((float)a1[0], (float)a1[1]));
    float m1 = fmaxf(fmaxf((float)b0[0], (float)b0[1]),
                     fmaxf((float)b1[0], (float)b1[1]));
    h2f st; st[0] = (_Float16)m0; st[1] = (_Float16)m1;
    *(h2f*)(out + o2) = st;
  }
}

// ------------------------- conv2 + pool -----------------------------------
// per-image implicit GEMM: M=64 (8x8), N=64, K=800 = 25 taps x 32 cin.
// Weight tile [64][32] per tap streamed into double-buffered LDS (TDM when
// available), one barrier per K-step.
__global__ __launch_bounds__(256)
void k_conv2(const _Float16* __restrict__ h1, const _Float16* __restrict__ W2t,
             const float* __restrict__ b2, _Float16* __restrict__ emb1h) {
  __shared__ _Float16 in_s[12 * 12 * 32];
  __shared__ _Float16 bt[2][64 * 32];
  __shared__ _Float16 co[64 * 64];          // [channel][position]
  const int imgId  = blockIdx.x;
  const int tid    = threadIdx.x;
  const int lane   = tid & 31;
  const int wave   = tid >> 5;
  const int laneHi = lane >> 4;
  const int laneLo = lane & 15;
  const int mt     = wave & 3;
  const int nt0    = (wave >> 2) * 2;

  {
    const unsigned int* s = (const unsigned int*)(h1 + (size_t)imgId * 4608);
    unsigned int* d = (unsigned int*)in_s;
    for (int i = tid; i < 2304; i += 256) d[i] = s[i];
  }
#if HAVE_TDM
  if (wave == 0) {
    tdm_load_f16_1d((unsigned int)(uintptr_t)&bt[0][0], W2t, 2048);
    __builtin_amdgcn_s_wait_tensorcnt(0);
  }
#else
  {
    const unsigned int* s = (const unsigned int*)W2t;
    unsigned int* d = (unsigned int*)&bt[0][0];
#pragma unroll
    for (int j = 0; j < 4; ++j) d[tid + j * 256] = s[tid + j * 256];
  }
#endif
  __syncthreads();

  const int p  = mt * 16 + laneLo;
  const int oy = p >> 3, ox = p & 7;
  v8f c0 = {}, c1 = {};

  for (int kt = 0; kt < 25; ++kt) {
    const _Float16* cur = &bt[kt & 1][0];
    // stream next tap's weight tile into the other buffer
#if HAVE_TDM
    if (wave == 0 && kt + 1 < 25)
      tdm_load_f16_1d((unsigned int)(uintptr_t)&bt[(kt + 1) & 1][0],
                      W2t + (kt + 1) * 2048, 2048);
#else
    if (kt + 1 < 25) {
      const unsigned int* s = (const unsigned int*)(W2t + (kt + 1) * 2048);
      unsigned int* d = (unsigned int*)&bt[(kt + 1) & 1][0];
#pragma unroll
      for (int j = 0; j < 4; ++j) d[tid + j * 256] = s[tid + j * 256];
      __builtin_prefetch(W2t + (kt + 2) * 2048, 0, 1);
    }
#endif
    const int ky = kt / 5, kx = kt - 5 * ky;
    v16h a  = load_frag(&in_s[((oy + ky) * 12 + (ox + kx)) * 32], laneHi);
    v16h b0 = load_frag(cur + (nt0 * 16 + laneLo) * 32, laneHi);
    v16h b1 = load_frag(cur + ((nt0 + 1) * 16 + laneLo) * 32, laneHi);
    c0 = wmma16(a, b0, c0);
    c1 = wmma16(a, b1, c1);
#if HAVE_TDM
    if (wave == 0 && kt + 1 < 25) __builtin_amdgcn_s_wait_tensorcnt(0);
#endif
    __syncthreads();
  }

  {
    int ch0 = nt0 * 16 + laneLo;
    int ch1 = ch0 + 16;
    float bb0 = b2[ch0], bb1 = b2[ch1];
    int row0 = mt * 16 + laneHi * 8;
    h8 s0, s1;
#pragma unroll
    for (int r = 0; r < 8; ++r) {
      s0[r] = (_Float16)fmaxf(c0[r] + bb0, 0.f);
      s1[r] = (_Float16)fmaxf(c1[r] + bb1, 0.f);
    }
    *(h8*)(&co[ch0 * 64 + row0]) = s0;      // ds_store_b128
    *(h8*)(&co[ch1 * 64 + row0]) = s1;      // ds_store_b128
  }
  __syncthreads();

  // 2x2 maxpool: 8x8x64 -> 4x4x64 -> flattened D=1024; channel pairs
  _Float16* out = emb1h + (size_t)imgId * 1024;
  for (int o2 = tid * 2; o2 < 1024; o2 += 512) {
    int ch = o2 & 63, pix = o2 >> 6;
    int px = pix & 3, py = pix >> 2;
    int p0 = (py * 2) * 8 + px * 2;
    h2f a0 = *(const h2f*)(co + ch * 64 + p0);
    h2f a1 = *(const h2f*)(co + ch * 64 + p0 + 8);
    h2f b0 = *(const h2f*)(co + (ch + 1) * 64 + p0);
    h2f b1 = *(const h2f*)(co + (ch + 1) * 64 + p0 + 8);
    float m0 = fmaxf(fmaxf((float)a0[0], (float)a0[1]),
                     fmaxf((float)a1[0], (float)a1[1]));
    float m1 = fmaxf(fmaxf((float)b0[0], (float)b0[1]),
                     fmaxf((float)b1[0], (float)b1[1]));
    h2f st; st[0] = (_Float16)m0; st[1] = (_Float16)m1;
    *(h2f*)(out + o2) = st;
  }
}

// ------------------------- attention scores -------------------------------
// s[row] = sigmoid( tanh(emb@W + b) . v + vb ), 16 rows per workgroup.
// A tile (16x1024 f16 = 32KB) staged to LDS (TDM when available); B read
// from transposed weights with b128 loads.  Epilogue pre-reduces across
// laneLo with shfl_xor before the per-row LDS atomic.
__global__ __launch_bounds__(256)
void k_att(const _Float16* __restrict__ emb, const _Float16* __restrict__ AWt,
           const float* __restrict__ b, const float* __restrict__ v,
           const float* __restrict__ vb, float* __restrict__ sOut) {
  __shared__ _Float16 at[16 * 1024];
  __shared__ float rowacc[16];
  const int tid    = threadIdx.x;
  const int lane   = tid & 31;
  const int wave   = tid >> 5;
  const int laneHi = lane >> 4;
  const int laneLo = lane & 15;
  const int nt0    = wave * 2;

  const _Float16* src = emb + (size_t)blockIdx.x * (16 * 1024);
#if HAVE_TDM
  if (wave == 0) {
    tdm_load_f16_1d((unsigned int)(uintptr_t)&at[0], src, 16 * 1024);
    __builtin_amdgcn_s_wait_tensorcnt(0);
  }
#else
  {
    const unsigned int* s = (const unsigned int*)src;
    unsigned int* d = (unsigned int*)at;
    for (int i = tid; i < 8192; i += 256) d[i] = s[i];
  }
#endif
  if (tid < 16) rowacc[tid] = 0.f;
  __syncthreads();

  v8f c0 = {}, c1 = {};
  for (int kt = 0; kt < 32; ++kt) {
    v16h a  = load_frag(&at[laneLo * 1024 + kt * 32], laneHi);
    v16h bv0 = load_frag(AWt + (size_t)(nt0 * 16 + laneLo) * 1024 + kt * 32, laneHi);
    v16h bv1 = load_frag(AWt + (size_t)((nt0 + 1) * 16 + laneLo) * 1024 + kt * 32, laneHi);
    c0 = wmma16(a, bv0, c0);
    c1 = wmma16(a, bv1, c1);
  }

  const int n0 = nt0 * 16 + laneLo;
  const int n1 = n0 + 16;
  const float bn0 = b[n0], vn0 = v[n0];
  const float bn1 = b[n1], vn1 = v[n1];
#pragma unroll
  for (int r = 0; r < 8; ++r) {
    float t = tanhf(c0[r] + bn0) * vn0 + tanhf(c1[r] + bn1) * vn1;
    t += __shfl_xor(t, 1, 32);              // reduce across laneLo (16 lanes)
    t += __shfl_xor(t, 2, 32);
    t += __shfl_xor(t, 4, 32);
    t += __shfl_xor(t, 8, 32);
    if (laneLo == 0) atomicAdd(&rowacc[r + laneHi * 8], t);   // ds_add_f32
  }
  __syncthreads();
  if (tid < 16) {
    float sc = rowacc[tid] + vb[0];
    sOut[blockIdx.x * 16 + tid] = 1.f / (1.f + expf(-sc));
  }
}

// ------------------------- softmax normalizer -----------------------------

__global__ void k_sumexp(const float* __restrict__ s, int n,
                         float* __restrict__ partials) {
  __shared__ float red[256];
  int tid = threadIdx.x;
  float acc = 0.f;
  for (int i = blockIdx.x * 256 + tid; i < n; i += gridDim.x * 256)
    acc += expf(s[i]);
  red[tid] = acc;
  __syncthreads();
  for (int off = 128; off > 0; off >>= 1) {
    if (tid < off) red[tid] += red[tid + off];
    __syncthreads();
  }
  if (tid == 0) partials[blockIdx.x] = red[0];
}

__global__ void k_sumfin(const float* __restrict__ partials, int n,
                         float* __restrict__ out) {
  if (threadIdx.x == 0) {
    float acc = 0.f;
    for (int i = 0; i < n; ++i) acc += partials[i];
    out[0] = acc;
  }
}

// ------------------------- weighted segment sum ---------------------------
// contiguous bags of 16 rows: out[bag] = sum_i softmax(s)[row] * emb[row]
__global__ __launch_bounds__(256)
void k_segsum(const _Float16* __restrict__ embIn, const float* __restrict__ s,
              const float* __restrict__ sumExp, float* __restrict__ outF,
              _Float16* __restrict__ outH) {
  const int bag = blockIdx.x;
  const int tid = threadIdx.x;
  const float inv = 1.f / sumExp[0];
  const int d0 = tid * 4;
  float acc[4] = {0.f, 0.f, 0.f, 0.f};
  for (int i = 0; i < 16; ++i) {
    int row = bag * 16 + i;
    float w = expf(s[row]) * inv;
    const _Float16* e = embIn + (size_t)row * 1024 + d0;
#pragma unroll
    for (int j = 0; j < 4; ++j) acc[j] += w * (float)e[j];
  }
#pragma unroll
  for (int j = 0; j < 4; ++j) {
    outF[(size_t)bag * 1024 + d0 + j] = acc[j];
    outH[(size_t)bag * 1024 + d0 + j] = (_Float16)acc[j];
  }
}

// ------------------------- level-3 + classifier ---------------------------

__global__ __launch_bounds__(256)
void k_final(const float* __restrict__ emb3f, const float* __restrict__ s3,
             const float* __restrict__ clsW, const float* __restrict__ clsB,
             const float* __restrict__ outW, const float* __restrict__ outB,
             float* __restrict__ dOut) {
  __shared__ float w3[64];
  __shared__ float outer[1024];
  __shared__ float y[512];
  __shared__ float red[256];
  __shared__ float ssum;
  const int tid = threadIdx.x;

  if (tid < 64) w3[tid] = expf(s3[tid]);
  __syncthreads();
  if (tid == 0) {
    float acc = 0.f;
    for (int i = 0; i < 64; ++i) acc += w3[i];
    ssum = acc;
  }
  __syncthreads();
  const float inv = 1.f / ssum;
  for (int d = tid; d < 1024; d += 256) {
    float acc = 0.f;
    for (int bb = 0; bb < 64; ++bb) acc += w3[bb] * emb3f[bb * 1024 + d];
    outer[d] = acc * inv;
  }
  __syncthreads();
  for (int j = tid; j < 512; j += 256) {
    float acc = clsB[j];
    for (int d = 0; d < 1024; ++d) acc += outer[d] * clsW[d * 512 + j];
    y[j] = acc;
  }
  __syncthreads();
  red[tid] = y[tid] * outW[tid] + y[tid + 256] * outW[tid + 256];
  __syncthreads();
  for (int off = 128; off > 0; off >>= 1) {
    if (tid < off) red[tid] += red[tid + off];
    __syncthreads();
  }
  if (tid == 0) dOut[0] = 1.f / (1.f + expf(-(red[0] + outB[0])));
}

// ---------------------------------------------------------------------------

extern "C" void kernel_launch(void* const* d_in, const int* in_sizes, int n_in,
                              void* d_out, int out_size, void* d_ws, size_t ws_size,
                              hipStream_t stream) {
  const float* x       = (const float*)d_in[0];
  const float* conv1_w = (const float*)d_in[1];
  const float* conv1_b = (const float*)d_in[2];
  const float* conv2_w = (const float*)d_in[3];
  const float* conv2_b = (const float*)d_in[4];
  const float* att1_W  = (const float*)d_in[5];
  const float* att1_b  = (const float*)d_in[6];
  const float* att1_v  = (const float*)d_in[7];
  const float* att1_vb = (const float*)d_in[8];
  const float* att2_W  = (const float*)d_in[9];
  const float* att2_b  = (const float*)d_in[10];
  const float* att2_v  = (const float*)d_in[11];
  const float* att2_vb = (const float*)d_in[12];
  const float* att3_W  = (const float*)d_in[13];
  const float* att3_b  = (const float*)d_in[14];
  const float* att3_v  = (const float*)d_in[15];
  const float* att3_vb = (const float*)d_in[16];
  const float* cls_W   = (const float*)d_in[17];
  const float* cls_b   = (const float*)d_in[18];
  const float* out_W   = (const float*)d_in[19];
  const float* out_b   = (const float*)d_in[20];
  // d_in[21]/[22]: bag ids — contiguous 16:1 by construction; not needed.

  // ---- workspace layout (256B aligned) ----
  char* base = (char*)d_ws;
  size_t off = 0;
  auto take = [&](size_t bytes) -> char* {
    char* p = base + off;
    off += (bytes + 255) & ~(size_t)255;
    return p;
  };
  _Float16* W1t    = (_Float16*)take(32 * 32 * 2);
  _Float16* W2t    = (_Float16*)take((800 + 64) * 64 * 2);  // +2 taps slack for prefetch
  _Float16* A1Wt   = (_Float16*)take((size_t)256 * 1024 * 2);
  _Float16* A2Wt   = (_Float16*)take((size_t)256 * 1024 * 2);
  _Float16* A3Wt   = (_Float16*)take((size_t)256 * 1024 * 2);
  _Float16* h1     = (_Float16*)take((size_t)N_INST * 4608 * 2);
  _Float16* emb1h  = (_Float16*)take((size_t)N_INST * 1024 * 2);
  float*    s1     = (float*)take((size_t)N_INST * 4);
  float*    s2     = (float*)take(NBAG1 * 4);
  float*    s3     = (float*)take(NBAG2 * 4);
  float*    emb2f  = (float*)take((size_t)NBAG1 * 1024 * 4);
  _Float16* emb2h  = (_Float16*)take((size_t)NBAG1 * 1024 * 2);
  float*    emb3f  = (float*)take((size_t)NBAG2 * 1024 * 4);
  _Float16* emb3h  = (_Float16*)take((size_t)NBAG2 * 1024 * 2);
  float*    part1  = (float*)take(64 * 4);
  float*    part2  = (float*)take(4 * 4);
  float*    accum  = (float*)take(2 * 4);

  // ---- weight packing (f32 -> transposed f16) ----
  k_pack_w1<<<1, 1024, 0, stream>>>(conv1_w, W1t);
  k_pack_w2<<<200, 256, 0, stream>>>(conv2_w, W2t);
  k_pack_att<<<1024, 256, 0, stream>>>(att1_W, A1Wt);
  k_pack_att<<<1024, 256, 0, stream>>>(att2_W, A2Wt);
  k_pack_att<<<1024, 256, 0, stream>>>(att3_W, A3Wt);

  // ---- backbone ----
  k_conv1<<<N_INST, 256, 0, stream>>>(x, W1t, conv1_b, h1);
  k_conv2<<<N_INST, 256, 0, stream>>>(h1, W2t, conv2_b, emb1h);

  // ---- level 1: instances -> B1 bags ----
  k_att<<<N_INST / 16, 256, 0, stream>>>(emb1h, A1Wt, att1_b, att1_v, att1_vb, s1);
  k_sumexp<<<64, 256, 0, stream>>>(s1, N_INST, part1);
  k_sumfin<<<1, 32, 0, stream>>>(part1, 64, accum + 0);
  k_segsum<<<NBAG1, 256, 0, stream>>>(emb1h, s1, accum + 0, emb2f, emb2h);

  // ---- level 2: B1 -> B2 bags ----
  k_att<<<NBAG1 / 16, 256, 0, stream>>>(emb2h, A2Wt, att2_b, att2_v, att2_vb, s2);
  k_sumexp<<<4, 256, 0, stream>>>(s2, NBAG1, part2);
  k_sumfin<<<1, 32, 0, stream>>>(part2, 4, accum + 1);
  k_segsum<<<NBAG2, 256, 0, stream>>>(emb2h, s2, accum + 1, emb3f, emb3h);

  // ---- level 3 scores + outer bag + classifier ----
  k_att<<<NBAG2 / 16, 256, 0, stream>>>(emb3h, A3Wt, att3_b, att3_v, att3_vb, s3);
  k_final<<<1, 256, 0, stream>>>(emb3f, s3, cls_W, cls_b, out_W, out_b, (float*)d_out);
}
